// TransfomerAttentionLayer_20555713478854
// MI455X (gfx1250) — matrix operand order repaired and checked by reading, not compiled, not run on hardware
//
#include <hip/hip_runtime.h>
#include <hip/hip_bf16.h>
#include <stdint.h>

// ---------------------------------------------------------------------------
// CDNA5 / gfx1250 WMMA helpers (wave32, v_wmma_f32_16x16x32_bf16)
// ---------------------------------------------------------------------------
typedef __attribute__((ext_vector_type(16))) __bf16 v16bf;
typedef __attribute__((ext_vector_type(8)))  float  v8f;

union V16U { v16bf v; uint4 q[2]; };

__device__ __forceinline__ __bf16 f2bf(float x) {
  unsigned u = __float_as_uint(x);
  u += 0x7FFFu + ((u >> 16) & 1u);          // round-to-nearest-even
  unsigned short s = (unsigned short)(u >> 16);
  return __builtin_bit_cast(__bf16, s);
}
__device__ __forceinline__ float bf2f(__bf16 b) {
  unsigned short s = __builtin_bit_cast(unsigned short, b);
  return __uint_as_float(((unsigned)s) << 16);
}
__device__ __forceinline__ unsigned pk2(float a, float b) {
  return (unsigned)__builtin_bit_cast(unsigned short, f2bf(a)) |
         ((unsigned)__builtin_bit_cast(unsigned short, f2bf(b)) << 16);
}
// monotone float<->uint encoding for atomicMax-based segment max
__device__ __forceinline__ unsigned encf(float x) {
  unsigned u = __float_as_uint(x);
  return (u & 0x80000000u) ? ~u : (u | 0x80000000u);
}
__device__ __forceinline__ float decf(unsigned u) {
  return (u & 0x80000000u) ? __uint_as_float(u & 0x7FFFFFFFu)
                           : __uint_as_float(~u);
}

// A fragment (16x32 bf16, M=rows of tile) from row-major LDS tile.
// ISA 7.12.2: lane l<16 -> row l, K={kb..kb+7, kb+16..kb+23};
//             lane l+16 -> row l, K={kb+8..kb+15, kb+24..kb+31}.
__device__ __forceinline__ v16bf load_A_lds(const __bf16* row0, int rowStride,
                                            int kbase, int lane) {
  int r = lane & 15, h = (lane >> 4) & 1;
  const __bf16* p = row0 + r * rowStride + kbase + 8 * h;
  V16U a;
  a.q[0] = *(const uint4*)(p);
  a.q[1] = *(const uint4*)(p + 16);
  return a.v;
}
// B fragment (32x16 bf16, K x N) where B(k,n) = W[n][k], W row-major bf16.
// lanes 0-15 hold K={kb..kb+7, kb+16..kb+23}, lanes 16-31 hold K={kb+8..kb+15, kb+24..kb+31}.
__device__ __forceinline__ v16bf load_B_glb(const __bf16* W, int rowStride,
                                            int ntile, int kbase, int lane) {
  int n  = ntile * 16 + (lane & 15);
  int k0 = kbase + ((lane >> 4) & 1) * 8;
  const __bf16* p = W + (size_t)n * rowStride + k0;
  V16U b;
  b.q[0] = *(const uint4*)(p);
  b.q[1] = *(const uint4*)(p + 16);
  return b.v;
}
#define WMMA_BF16(A, B, C) \
  __builtin_amdgcn_wmma_f32_16x16x32_bf16(false, (A), false, (B), (short)0, (C), false, false)

// ---------------------------------------------------------------------------
// Kernel 0: fp32 -> bf16 weight conversion
// ---------------------------------------------------------------------------
__global__ void k_cvt(const float* __restrict__ src, __bf16* __restrict__ dst, int n) {
  int t = blockIdx.x * blockDim.x + threadIdx.x;
  if (t < n) dst[t] = f2bf(src[t]);
}

// ---------------------------------------------------------------------------
// Kernel 1: Q_nodes = [h_dst | cos(time_bias)] @ Wq^T + bq   (Nd x 128)
// one wave per 16-row tile, K=256 -> 8 k-steps x 8 n-tiles WMMAs
// ---------------------------------------------------------------------------
__global__ __launch_bounds__(128) void k_qnodes(
    const float* __restrict__ h, const float* __restrict__ tbias,
    const __bf16* __restrict__ Wq, const float* __restrict__ bq,
    float* __restrict__ Qn, int Nd) {
  __shared__ __align__(16) __bf16 aT[4][16][256];
  int lane = threadIdx.x & 31, w = threadIdx.x >> 5;
  int tile = blockIdx.x * 4 + w;
  if (tile * 16 >= Nd) return;
  int base = tile * 16;

  const float4* tv = (const float4*)tbias;
  float4 t = tv[lane];
  unsigned tz0 = pk2(__cosf(t.x), __cosf(t.y));
  unsigned tz1 = pk2(__cosf(t.z), __cosf(t.w));
  for (int r = 0; r < 16; ++r) {
    const float4* hv = (const float4*)(h + (size_t)(base + r) * 128);
    float4 x = hv[lane];
    *(uint2*)&aT[w][r][lane * 4]       = make_uint2(pk2(x.x, x.y), pk2(x.z, x.w));
    *(uint2*)&aT[w][r][128 + lane * 4] = make_uint2(tz0, tz1);
  }
  int half = (lane >> 4) & 1;
  for (int nt = 0; nt < 8; ++nt) {
    v8f acc = {};
    for (int ks = 0; ks < 8; ++ks) {
      v16bf A = load_A_lds(&aT[w][0][0], 256, ks * 32, lane);
      v16bf B = load_B_glb(Wq, 256, nt, ks * 32, lane);
      acc = WMMA_BF16(A, B, acc);
    }
    int col = nt * 16 + (lane & 15);
    float bias = bq[col];
    for (int r = 0; r < 8; ++r)
      Qn[(size_t)(base + half * 8 + r) * 128 + col] = acc[r] + bias;
  }
}

// ---------------------------------------------------------------------------
// Kernel 2: per 32-edge tile (two 16-row A tiles sharing each B fragment):
// K,V = [h_src|f|cos(dt*freq+bias)] @ {Wk,Wv}^T + b
// then logits = leakyrelu(sum_head(Q*K)), store logits + V(bf16), seg-max.
// Head h == N-tile h (16 features); C layout puts one feature per lane, so the
// per-head dot is a 16-lane shfl_xor reduction over the accumulator regs.
// ---------------------------------------------------------------------------
__global__ __launch_bounds__(128) void k_kv(
    const float* __restrict__ h, const float* __restrict__ f,
    const float* __restrict__ dt, const int* __restrict__ dst_idx,
    const float* __restrict__ freq, const float* __restrict__ tbias,
    const __bf16* __restrict__ Wk, const __bf16* __restrict__ Wv,
    const float* __restrict__ bk, const float* __restrict__ bv,
    const float* __restrict__ Qn,
    float* __restrict__ logits, unsigned* __restrict__ segmax,
    __bf16* __restrict__ Vbf, int Nd, int E) {
  __shared__ __align__(16) __bf16 kvT[4][32][384];   // 96 KB / block
  int lane = threadIdx.x & 31, w = threadIdx.x >> 5;
  int tile = blockIdx.x * 4 + w;
  if (tile * 32 >= E) return;
  int base = tile * 32;

  const float4* fv = (const float4*)freq;
  const float4* bv4 = (const float4*)tbias;
  float4 fq = fv[lane], tb = bv4[lane];
  for (int e = 0; e < 32; ++e) {
    int edge = base + e;
    const float4* hs = (const float4*)(h + (size_t)(Nd + edge) * 128);
    const float4* fr = (const float4*)(f + (size_t)edge * 128);
    float4 x = hs[lane];
    *(uint2*)&kvT[w][e][lane * 4] = make_uint2(pk2(x.x, x.y), pk2(x.z, x.w));
    float4 y = fr[lane];
    *(uint2*)&kvT[w][e][128 + lane * 4] = make_uint2(pk2(y.x, y.y), pk2(y.z, y.w));
    float dtv = dt[edge];
    *(uint2*)&kvT[w][e][256 + lane * 4] = make_uint2(
        pk2(__cosf(dtv * fq.x + tb.x), __cosf(dtv * fq.y + tb.y)),
        pk2(__cosf(dtv * fq.z + tb.z), __cosf(dtv * fq.w + tb.w)));
  }
  int half = (lane >> 4) & 1;
  int dsts[2][8];
  for (int tsub = 0; tsub < 2; ++tsub)
    for (int r = 0; r < 8; ++r)
      dsts[tsub][r] = dst_idx[base + tsub * 16 + half * 8 + r];

  for (int nt = 0; nt < 8; ++nt) {
    v8f ak0 = {}, av0 = {}, ak1 = {}, av1 = {};
    for (int ks = 0; ks < 12; ++ks) {          // K = 384
      v16bf A0 = load_A_lds(&kvT[w][0][0],  384, ks * 32, lane);
      v16bf A1 = load_A_lds(&kvT[w][16][0], 384, ks * 32, lane);
      v16bf Bk = load_B_glb(Wk, 384, nt, ks * 32, lane);
      v16bf Bv = load_B_glb(Wv, 384, nt, ks * 32, lane);
      ak0 = WMMA_BF16(A0, Bk, ak0);
      ak1 = WMMA_BF16(A1, Bk, ak1);
      av0 = WMMA_BF16(A0, Bv, av0);
      av1 = WMMA_BF16(A1, Bv, av1);
    }
    int col = nt * 16 + (lane & 15);
    float bkc = bk[col], bvc = bv[col];
    for (int tsub = 0; tsub < 2; ++tsub) {
      const v8f& ak = tsub ? ak1 : ak0;
      const v8f& av = tsub ? av1 : av0;
      for (int r = 0; r < 8; ++r) {
        int edge = base + tsub * 16 + half * 8 + r;
        Vbf[(size_t)edge * 128 + col] = f2bf(av[r] + bvc);
        float q = Qn[(size_t)dsts[tsub][r] * 128 + col];
        float p = q * (ak[r] + bkc);
        p += __shfl_xor(p, 1, 16);
        p += __shfl_xor(p, 2, 16);
        p += __shfl_xor(p, 4, 16);
        p += __shfl_xor(p, 8, 16);
        if ((lane & 15) == 0) {
          float lg = p > 0.f ? p : 0.2f * p;     // leaky_relu(0.2)
          logits[(size_t)edge * 8 + nt] = lg;
          atomicMax(&segmax[(size_t)dsts[tsub][r] * 8 + nt], encf(lg));
        }
      }
    }
  }
}

// ---------------------------------------------------------------------------
// Kernel 3: ex = exp(logit - segmax[dst]); seg_sum += ex
// ---------------------------------------------------------------------------
__global__ void k_expsum(const int* __restrict__ dst_idx,
                         const unsigned* __restrict__ segmax,
                         float* __restrict__ logits,
                         float* __restrict__ segsum, int E) {
  int t = blockIdx.x * blockDim.x + threadIdx.x;
  if (t >= E * 8) return;
  int e = t >> 3, hh = t & 7;
  int dst = dst_idx[e];
  float m = decf(segmax[(size_t)dst * 8 + hh]);
  float ex = __expf(logits[t] - m);
  logits[t] = ex;
  unsafeAtomicAdd(&segsum[(size_t)dst * 8 + hh], ex);   // hw global_atomic_add_f32
}

// ---------------------------------------------------------------------------
// Kernel 4: agg[dst] += V[e] * (ex / segsum[dst]); thread = (edge, 4 feats)
// ---------------------------------------------------------------------------
__global__ void k_agg(const int* __restrict__ dst_idx,
                      const float* __restrict__ logits,
                      const float* __restrict__ segsum,
                      const __bf16* __restrict__ Vbf,
                      float* __restrict__ agg, int E) {
  long long t = (long long)blockIdx.x * blockDim.x + threadIdx.x;
  if (t >= (long long)E * 32) return;
  int e = (int)(t >> 5);
  int c0 = ((int)t & 31) * 4;        // 4 feats, all in one head (head = c0/16)
  int hh = c0 >> 4;
  int dst = dst_idx[e];
  float att = logits[(size_t)e * 8 + hh] / segsum[(size_t)dst * 8 + hh];
  const __bf16* vp = Vbf + (size_t)e * 128 + c0;
  float* ap = agg + (size_t)dst * 128 + c0;
  unsafeAtomicAdd(ap + 0, bf2f(vp[0]) * att);
  unsafeAtomicAdd(ap + 1, bf2f(vp[1]) * att);
  unsafeAtomicAdd(ap + 2, bf2f(vp[2]) * att);
  unsafeAtomicAdd(ap + 3, bf2f(vp[3]) * att);
}

// ---------------------------------------------------------------------------
// Kernel 5: rst = relu([agg | h_dst] @ Wout^T + bout); layernorm -> out
// ---------------------------------------------------------------------------
__global__ __launch_bounds__(128) void k_out(
    const float* __restrict__ h, const float* __restrict__ agg,
    const __bf16* __restrict__ Wout, const float* __restrict__ bout,
    const float* __restrict__ gamma, const float* __restrict__ beta,
    float* __restrict__ out, int Nd) {
  __shared__ __align__(16) __bf16 aT[4][16][256];
  __shared__ float res[4][16][128];
  int lane = threadIdx.x & 31, w = threadIdx.x >> 5;
  int tile = blockIdx.x * 4 + w;
  if (tile * 16 >= Nd) return;
  int base = tile * 16;

  for (int r = 0; r < 16; ++r) {
    const float4* av = (const float4*)(agg + (size_t)(base + r) * 128);
    float4 x = av[lane];
    *(uint2*)&aT[w][r][lane * 4] = make_uint2(pk2(x.x, x.y), pk2(x.z, x.w));
    const float4* hv = (const float4*)(h + (size_t)(base + r) * 128);
    float4 y = hv[lane];
    *(uint2*)&aT[w][r][128 + lane * 4] = make_uint2(pk2(y.x, y.y), pk2(y.z, y.w));
  }
  int half = (lane >> 4) & 1;
  for (int nt = 0; nt < 8; ++nt) {
    v8f acc = {};
    for (int ks = 0; ks < 8; ++ks) {
      v16bf A = load_A_lds(&aT[w][0][0], 256, ks * 32, lane);
      v16bf B = load_B_glb(Wout, 256, nt, ks * 32, lane);
      acc = WMMA_BF16(A, B, acc);
    }
    int col = nt * 16 + (lane & 15);
    float b = bout[col];
    for (int r = 0; r < 8; ++r) {
      float x = acc[r] + b;
      res[w][half * 8 + r][col] = x > 0.f ? x : 0.f;    // relu
    }
  }
  // layernorm: lane pair (l, l+16) splits the 128 cols of row l&15
  int row = lane & 15;
  float s1 = 0.f, s2 = 0.f;
  for (int c = half * 64; c < half * 64 + 64; ++c) {
    float x = res[w][row][c];
    s1 += x; s2 += x * x;
  }
  s1 += __shfl_xor(s1, 16, 32);
  s2 += __shfl_xor(s2, 16, 32);
  float mu  = s1 * (1.f / 128.f);
  float var = s2 * (1.f / 128.f) - mu * mu;
  float inv = rsqrtf(var + 1e-5f);
  for (int c = half * 64; c < half * 64 + 64; ++c) {
    out[(size_t)(base + row) * 128 + c] =
        (res[w][row][c] - mu) * inv * gamma[c] + beta[c];
  }
}

// ---------------------------------------------------------------------------
// Launcher
// ---------------------------------------------------------------------------
extern "C" void kernel_launch(void* const* d_in, const int* in_sizes, int n_in,
                              void* d_out, int out_size, void* d_ws, size_t ws_size,
                              hipStream_t stream) {
  const float* h      = (const float*)d_in[0];
  const float* f      = (const float*)d_in[1];
  const float* dt     = (const float*)d_in[2];
  const int*   dsti   = (const int*)d_in[3];
  const float* freq   = (const float*)d_in[4];
  const float* tbias  = (const float*)d_in[5];
  const float* Wq     = (const float*)d_in[6];
  const float* bq     = (const float*)d_in[7];
  const float* Wk     = (const float*)d_in[8];
  const float* bk     = (const float*)d_in[9];
  const float* Wv     = (const float*)d_in[10];
  const float* bv     = (const float*)d_in[11];
  const float* Wout   = (const float*)d_in[12];
  const float* bout   = (const float*)d_in[13];
  const float* gamma  = (const float*)d_in[14];
  const float* beta   = (const float*)d_in[15];
  (void)n_in; (void)out_size; (void)ws_size;

  const int E  = in_sizes[2];                 // dt has E elements
  const int Nd = in_sizes[0] / 128 - E;       // h has (Nd+E)*128

  char* ws = (char*)d_ws;
  size_t off = 0;
  auto carve = [&](size_t bytes) {
    char* p = ws + off;
    off = (off + bytes + 255) & ~(size_t)255;
    return p;
  };
  __bf16*   wq_bf  = (__bf16*)carve((size_t)128 * 256 * 2);
  __bf16*   wk_bf  = (__bf16*)carve((size_t)128 * 384 * 2);
  __bf16*   wv_bf  = (__bf16*)carve((size_t)128 * 384 * 2);
  __bf16*   wo_bf  = (__bf16*)carve((size_t)128 * 256 * 2);
  float*    Qn     = (float*)carve((size_t)Nd * 128 * 4);
  unsigned* segmax = (unsigned*)carve((size_t)Nd * 8 * 4);
  float*    segsum = (float*)carve((size_t)Nd * 8 * 4);
  float*    agg    = (float*)carve((size_t)Nd * 128 * 4);
  float*    logits = (float*)carve((size_t)E * 8 * 4);
  __bf16*   Vbf    = (__bf16*)carve((size_t)E * 128 * 2);

  // accumulator init (segmax: encoded order, 0 = minimum encoded value)
  hipMemsetAsync(segmax, 0, (size_t)Nd * 8 * 4, stream);
  hipMemsetAsync(segsum, 0, (size_t)Nd * 8 * 4, stream);
  hipMemsetAsync(agg,    0, (size_t)Nd * 128 * 4, stream);

  k_cvt<<<(128 * 256 + 255) / 256, 256, 0, stream>>>(Wq,   wq_bf, 128 * 256);
  k_cvt<<<(128 * 384 + 255) / 256, 256, 0, stream>>>(Wk,   wk_bf, 128 * 384);
  k_cvt<<<(128 * 384 + 255) / 256, 256, 0, stream>>>(Wv,   wv_bf, 128 * 384);
  k_cvt<<<(128 * 256 + 255) / 256, 256, 0, stream>>>(Wout, wo_bf, 128 * 256);

  k_qnodes<<<(Nd + 63) / 64, 128, 0, stream>>>(h, tbias, wq_bf, bq, Qn, Nd);
  k_kv<<<(E + 127) / 128, 128, 0, stream>>>(h, f, dt, dsti, freq, tbias,
                                            wk_bf, wv_bf, bk, bv, Qn,
                                            logits, segmax, Vbf, Nd, E);
  k_expsum<<<(int)(((long long)E * 8 + 255) / 256), 256, 0, stream>>>(
      dsti, segmax, logits, segsum, E);
  k_agg<<<(int)(((long long)E * 32 + 255) / 256), 256, 0, stream>>>(
      dsti, logits, segsum, Vbf, agg, E);
  k_out<<<(Nd + 63) / 64, 128, 0, stream>>>(h, agg, wo_bf, bout, gamma, beta,
                                            (float*)d_out, Nd);
}
